// Upsample_86801289052476
// MI455X (gfx1250) — matrix operand order repaired and checked by reading, not compiled, not run on hardware
//
#include <hip/hip_runtime.h>
#include <hip/hip_bf16.h>

typedef _Float16 v4h  __attribute__((ext_vector_type(4)));
typedef _Float16 v16h __attribute__((ext_vector_type(16)));
typedef float    v8f  __attribute__((ext_vector_type(8)));

// Problem constants (from reference)
#define BATCH 8
#define HW    128      // input H=W
#define CH    128      // input channels
#define OC    128      // output channels
#define OHW   256      // output H=W

// Weff fragment table: [phase(4)][tap(9)][octile(8)][kchunk(4)][lane(32)][16 f16]
// total = 4*9*8*4*32*16 = 589824 f16 = 1,179,648 bytes in d_ws
#define WEFF_ELEMS (4*9*8*4*32*16)

// Block spatial tile: 16 P-rows x 16 Q-cols.
// LDS input halo: 19 rows x 19 cols x 128 ch (f16), col stride padded to 136 f16
// (272B = 68 banks) so the 16 M-lanes of an A-fragment ds_load_b128 hit all 64 banks.
// Channels are stored permuted within each 32-group: [0..7,16..23,8..15,24..31],
// so each lane's A-fragment is ONE contiguous 32B chunk -> v16h load.
#define PROWS 16
#define HROWS (PROWS + 3)   // 19 halo rows: P0-1 .. P0+17
#define CSTR  136
#define RSTR  (19*CSTR)
#define LDS_ELEMS (HROWS*RSTR)

// Load a 32B per-lane fragment as one <16 x half> with 16B alignment
// (backend splits into two b128 loads into adjacent VGPRs).
__device__ __forceinline__ v16h ld_v16h(const _Float16* p) {
    struct __attribute__((packed, aligned(16))) W { v16h v; };
    return reinterpret_cast<const W*>(p)->v;
}

// ---------------------------------------------------------------------------
// Precompute the effective transposed-conv weights, already swizzled into
// WMMA B-fragment order.  Weff[py,px][ty,tx][c][oc] =
//   sum_{dy,dx} g[2ty+py-dy] * g[2tx+px-dx] * W[dy,dx,c,oc],  g=[1,3,3,1]/4
// B-fragment layout per (phase,tap,octile,kchunk): [lane(32)][16 f16], where
// lane l holds N = l%16 and K = (l/16)*8 + (j&7) + (j>>3)*16  (ISA 32x16 f16 B layout).
// ---------------------------------------------------------------------------
__global__ __launch_bounds__(256) void weff_precompute(
    const float* __restrict__ conv_w, _Float16* __restrict__ wf)
{
    int i = blockIdx.x * 256 + threadIdx.x;
    if (i >= WEFF_ELEMS) return;
    int e  = i & 15;
    int l  = (i >> 4) & 31;
    int kc = (i >> 9) & 3;
    int n  = (i >> 11) & 7;
    int t  = (i >> 14) % 9;
    int p4 = i / (9 * 16384);
    int lm = l & 15, hi = l >> 4;
    int oc = n * 16 + lm;
    int c  = kc * 32 + hi * 8 + (e & 7) + (e >> 3) * 16;
    int py = p4 >> 1, px = p4 & 1;
    int ty = t / 3,   tx = t % 3;
    const float g[4] = {0.25f, 0.75f, 0.75f, 0.25f};
    float acc = 0.f;
    for (int dy = 0; dy < 3; ++dy) {
        int ay = 2 * ty + py - dy;
        if (ay < 0 || ay > 3) continue;
        for (int dx = 0; dx < 3; ++dx) {
            int ax = 2 * tx + px - dx;
            if (ax < 0 || ax > 3) continue;
            acc += g[ay] * g[ax] * conv_w[((size_t)(dy * 3 + dx) * CH + c) * OC + oc];
        }
    }
    wf[i] = (_Float16)acc;
}

// ---------------------------------------------------------------------------
// Fused upsample+FIR+conv3x3 as 4-phase implicit GEMM with WMMA f16->f32.
// Block: 256 threads (8 waves). Block tile: 16 P-rows x 16 Q-cols, all 4
// phases, all 128 OC -> 512 KB of output. Wave w owns OC tile [16w, 16w+16).
// Each B fragment load is amortized over 16 WMMAs (16 independent
// accumulators, one per P-row) -> total L2 weight traffic ~0.6 GB,
// comfortably under the L2 ceiling vs the 320 MB / 13.7 us HBM floor,
// and the 16-way independent accumulator chain keeps the matrix pipe full.
// ---------------------------------------------------------------------------
__global__ __launch_bounds__(256) void upconv_wmma(
    const float* __restrict__ x, const _Float16* __restrict__ wf,
    const float* __restrict__ bias, float* __restrict__ out)
{
    __shared__ _Float16 xs[LDS_ELEMS];

    const int Q0 = blockIdx.x * 16;
    const int P0 = blockIdx.y * PROWS;
    const int bb = blockIdx.z;
    const int tid = threadIdx.x;

    // ---- load input halo rows P0-1..P0+17, cols Q0-1..Q0+17, f32 -> f16 ----
    // Channel permutation within each 32-group: blocks of 8 reordered 0,2,1,3
    // so A-fragments become contiguous per lane.
    for (int u = tid; u < HROWS * 19 * (CH / 4); u += 256) {
        int c4  = u & 31;            // channel group of 4 (0..31)
        int col = (u >> 5) % 19;
        int row = u / (32 * 19);
        int gy = P0 - 1 + row;
        int gx = Q0 - 1 + col;
        float4 v = make_float4(0.f, 0.f, 0.f, 0.f);
        if (gy >= 0 && gy < HW && gx >= 0 && gx < HW)
            v = *(const float4*)(x + ((size_t)((bb * HW + gy) * HW + gx)) * CH + c4 * 4);
        int kcg = c4 >> 3;                       // 32-channel group
        int j   = (c4 >> 1) & 3;                 // 8-channel block in group
        int w   = c4 & 1;                        // 4-channel half of block
        int jp  = ((j & 1) << 1) | (j >> 1);     // permute 0,1,2,3 -> 0,2,1,3
        int s4  = kcg * 8 + jp * 2 + w;          // permuted 4-channel slot
        v4h hv = { (_Float16)v.x, (_Float16)v.y, (_Float16)v.z, (_Float16)v.w };
        *(v4h*)&xs[row * RSTR + col * CSTR + s4 * 4] = hv;
    }
    __syncthreads();

    const int wv   = tid >> 5;       // wave id = OC tile
    const int lane = tid & 31;
    const int lm   = lane & 15;
    const int hi   = lane >> 4;
    const int oc   = wv * 16 + lm;
    const float bv = bias[oc];

    const size_t outB = (size_t)bb * OHW * OHW * OC;

    #pragma unroll 1
    for (int p4 = 0; p4 < 4; ++p4) {
        const int py = p4 >> 1, px = p4 & 1;
        v8f acc[PROWS];
        #pragma unroll
        for (int r = 0; r < PROWS; ++r) acc[r] = (v8f){};

        #pragma unroll 1
        for (int ty = 0; ty < 3; ++ty) {
            #pragma unroll 1
            for (int tx = 0; tx < 3; ++tx) {
                // A base: 16 spatial positions (M = lm -> patch col lm+tx+px);
                // lane-group picks its contiguous 32B half of the 32-chunk.
                const _Float16* ap0 =
                    &xs[(ty + py) * RSTR + (tx + px + lm) * CSTR + hi * 16];
                // B: swizzled Weff chunk for (phase, tap, octile); 32B per lane.
                const _Float16* bp =
                    wf + ((size_t)((p4 * 9 + ty * 3 + tx) * 8 + wv) * 4) * 512
                       + lane * 16;
                #pragma unroll
                for (int kc = 0; kc < 4; ++kc) {
                    v16h b = ld_v16h(bp + kc * 512);
                    #pragma unroll
                    for (int r = 0; r < PROWS; ++r) {
                        v16h a = ld_v16h(ap0 + r * RSTR + kc * 32);
                        acc[r] = __builtin_amdgcn_wmma_f32_16x16x32_f16(
                                     false, a, false, b, (short)0, acc[r],
                                     false, false);
                    }
                }
            }
        }
        // ---- epilogue: D lane holds N=lm, VGPR v -> M = v + 8*hi ----
        #pragma unroll
        for (int r = 0; r < PROWS; ++r) {
            const int p = 2 * (P0 + r) + py;
            #pragma unroll
            for (int v = 0; v < 8; ++v) {
                int M = v + hi * 8;
                int q = 2 * (Q0 + M) + px;
                __builtin_nontemporal_store(
                    acc[r][v] + bv,
                    out + outB + ((size_t)p * OHW + q) * OC + oc);
            }
        }
    }
}

extern "C" void kernel_launch(void* const* d_in, const int* in_sizes, int n_in,
                              void* d_out, int out_size, void* d_ws, size_t ws_size,
                              hipStream_t stream) {
    const float* x      = (const float*)d_in[0];   // (8,128,128,128) f32
    const float* conv_w = (const float*)d_in[1];   // (3,3,128,128)   f32 HWIO
    const float* conv_b = (const float*)d_in[2];   // (128,)          f32
    float* out = (float*)d_out;                    // (8,256,256,128) f32
    _Float16* wf = (_Float16*)d_ws;                // 1.18 MB swizzled Weff

    weff_precompute<<<WEFF_ELEMS / 256, 256, 0, stream>>>(conv_w, wf);

    dim3 grid(HW / 16, HW / PROWS, BATCH);         // (8, 8, 8)
    upconv_wmma<<<grid, 256, 0, stream>>>(x, wf, conv_b, out);
}
